// NeuralSDE_BPTT_63221918597548
// MI455X (gfx1250) — matrix operand order repaired and checked by reading, not compiled, not run on hardware
//
#include <hip/hip_runtime.h>
#include <hip/hip_bf16.h>

// ---------------------------------------------------------------------------
// Neural SDE RK4 integrator for MI455X (gfx1250).
// B=256 batch -> 16 persistent workgroups, each owning a 16-row batch tile and
// iterating all T=1000 steps locally. All MLP weights live in LDS (f16,
// ~248KB of the 320KB/WGP), staged once in PRE-SWIZZLED WMMA B-fragment order
// so every B operand is one aligned 32-byte LDS load (2x ds_load_b128).
// GEMMs via v_wmma_f32_16x16x32_f16 (f16 in, f32 accumulate), K-loop fully
// unrolled (templated chunk count) so LDS loads pipeline across WMMAs.
// Narrow-output layers (L3, G1, G2) are split-K across all 16 waves with an
// LDS partial reduction to shorten the serial WMMA chain per step.
// Activations use branch-free v_exp/v_log/v_rcp based tanh & softplus.
// ---------------------------------------------------------------------------

typedef __attribute__((ext_vector_type(16))) _Float16 v16h;
typedef __attribute__((ext_vector_type(8)))  _Float16 v8h;
typedef __attribute__((ext_vector_type(8)))  float    v8f;

#define B_TOT   256
#define T_STEPS 1000
#define N_S     4
#define N_X     64
#define D_IN    68
#define KP      96      // D_IN padded to 3 WMMA K-chunks of 32
#define H_F     256
#define H_G     128
#define MT      16      // batch tile rows per workgroup
#define NTHREADS 512    // 16 wave32 waves
#define NWAVES  16
#define DT      0.01f
#define SQRT_DT 0.1f
#define SIGMA   0.1f

// ---- LDS layout (element counts) ----
#define E_WF1 (KP * H_F)        // 24576 f16
#define E_WF2 (H_F * H_F)       // 65536 f16
#define E_WF3 (H_F * N_X)       // 16384 f16
#define E_WG1 (KP * H_G)        // 12288 f16
#define E_WG2 (H_G * N_X)       //  8192 f16
#define E_Z   (MT * KP)         //  1536 f16
#define E_H1  (MT * H_F)        //  4096 f16
#define E_H2  (MT * H_F)        //  4096 f16
#define E_HG  (MT * H_G)        //  2048 f16
#define H_ELEMS (E_WF1+E_WF2+E_WF3+E_WG1+E_WG2+E_Z+E_H1+E_H2+E_HG)
// floats: sPart(16 waves x 256) + X,K,Acc,Gv + U + biases
#define F_ELEMS (NWAVES*256 + 4*MT*N_X + MT*N_S + H_F+H_F+N_X+H_G+N_X)
#define SMEM_BYTES ((size_t)H_ELEMS*2 + (size_t)F_ELEMS*4)

// ---------------------------------------------------------------------------
// Stage a (Ksrc x N) f32 weight into LDS as f16, zero-padded to Kpad rows,
// in WMMA B-fragment order: element (k,n) ->
//   kc=k/32, kk=k%32, nt=n/16, nn=n%16, lane=(kk/16)*16+nn, e=kk%16
//   sidx = (((nt*numKc + kc)*32 + lane)*16 + e)
// ---------------------------------------------------------------------------
__device__ __forceinline__ void stage_weight_sw(const float* __restrict__ W,
                                                _Float16* __restrict__ sW,
                                                int Ksrc, int Kpad, int N, int tid) {
  const int numKc = Kpad >> 5;
  for (int i = tid; i < Kpad * N; i += NTHREADS) {
    int k = i / N, n = i - k * N;
    float v = (k < Ksrc) ? W[k * N + n] : 0.f;
    int kc = k >> 5, kk = k & 31;
    int nt = n >> 4, nn = n & 15;
    int lane2 = ((kk >> 4) << 4) | nn;
    int e = kk & 15;
    sW[((((nt * numKc + kc) << 5) + lane2) << 4) + e] = (_Float16)v;
  }
}

// A (16x32 f16) fragment from row-major activations: lane L holds row M=L%16;
// its 16 elements are contiguous at K = half*16 .. half*16+15.
__device__ __forceinline__ v16h load_a_frag(const _Float16* __restrict__ A,
                                            int ldA, int k0, int lane) {
  const int m    = lane & 15;
  const int half = (lane >> 4) & 1;
  const _Float16* p = A + m * ldA + k0 + half * 16;
  v8h g0 = *(const v8h*)p;
  v8h g1 = *(const v8h*)(p + 8);
  return __builtin_shufflevector(g0, g1, 0,1,2,3,4,5,6,7,8,9,10,11,12,13,14,15);
}

// Partial 16x16 tile: A[16xK] x B-swizzled, K-chunks [kc0, kc0+NKC).
// NKC is a template constant -> fully unrolled, loads pipeline across WMMAs.
template <int NKC>
__device__ __forceinline__ v8f gemm_tile(const _Float16* __restrict__ A, int ldA,
                                         const _Float16* __restrict__ sW,
                                         int numKcTot, int nt, int kc0, int lane) {
  v8f c = {0.f,0.f,0.f,0.f,0.f,0.f,0.f,0.f};
  const _Float16* wp = sW + ((((size_t)nt * numKcTot + kc0) << 5) + lane) * 16;
#pragma unroll
  for (int kc = 0; kc < NKC; ++kc) {
    v16h a = load_a_frag(A, ldA, (kc0 + kc) << 5, lane);
    v16h b = *(const v16h*)(wp + (size_t)kc * 32 * 16); // 32B aligned
    c = __builtin_amdgcn_wmma_f32_16x16x32_f16(false, a, false, b,
                                               (short)0, c, false, false);
  }
  return c;
}

// Branch-free fast activations on hardware transcendentals.
__device__ __forceinline__ float tanh_fast(float x) {
  float e = __expf(2.0f * x);                           // v_exp_f32
  return 1.0f - 2.0f * __builtin_amdgcn_rcpf(e + 1.0f); // v_rcp_f32
}
__device__ __forceinline__ float softplus_fast(float x) {
  return fmaxf(x, 0.f) + __logf(1.f + __expf(-fabsf(x)));
}

__global__ __launch_bounds__(NTHREADS)
void sde_rk4_kernel(const float* __restrict__ Inputs,  // (B, 68, T)
                    const float* __restrict__ noise,   // (B, 64, T)
                    const float* __restrict__ Wf1, const float* __restrict__ bf1,
                    const float* __restrict__ Wf2, const float* __restrict__ bf2,
                    const float* __restrict__ Wf3, const float* __restrict__ bf3,
                    const float* __restrict__ Wg1, const float* __restrict__ bg1,
                    const float* __restrict__ Wg2, const float* __restrict__ bg2,
                    float* __restrict__ out)           // (B, 64, T)
{
  extern __shared__ unsigned char smem_raw[];
  _Float16* sWf1 = (_Float16*)smem_raw;
  _Float16* sWf2 = sWf1 + E_WF1;
  _Float16* sWf3 = sWf2 + E_WF2;
  _Float16* sWg1 = sWf3 + E_WF3;
  _Float16* sWg2 = sWg1 + E_WG1;
  _Float16* sZ   = sWg2 + E_WG2;
  _Float16* sH1  = sZ   + E_Z;
  _Float16* sH2  = sH1  + E_H1;
  _Float16* sHg  = sH2  + E_H2;
  float* sPart = (float*)(sHg + E_HG);    // 32B-aligned: H_ELEMS*2 % 32 == 0
  float* sX   = sPart + NWAVES * 256;
  float* sK   = sX   + MT * N_X;
  float* sAcc = sK   + MT * N_X;
  float* sGv  = sAcc + MT * N_X;
  float* sU   = sGv  + MT * N_X;          // MT x N_S
  float* sbf1 = sU   + MT * N_S;
  float* sbf2 = sbf1 + H_F;
  float* sbf3 = sbf2 + H_F;
  float* sbg1 = sbf3 + N_X;
  float* sbg2 = sbg1 + H_G;

  const int tid  = threadIdx.x;
  const int lane = tid & 31;
  const int wave = tid >> 5;              // 0..15
  const int b0   = blockIdx.x * MT;

  // ------------- one-time weight staging (f32 -> f16, swizzled) -------------
  stage_weight_sw(Wf1, sWf1, D_IN, KP,  H_F, tid);
  stage_weight_sw(Wf2, sWf2, H_F,  H_F, H_F, tid);
  stage_weight_sw(Wf3, sWf3, H_F,  H_F, N_X, tid);
  stage_weight_sw(Wg1, sWg1, D_IN, KP,  H_G, tid);
  stage_weight_sw(Wg2, sWg2, H_G,  H_G, N_X, tid);
  for (int i = tid; i < H_F; i += NTHREADS) { sbf1[i] = bf1[i]; sbf2[i] = bf2[i]; }
  for (int i = tid; i < H_G; i += NTHREADS) sbg1[i] = bg1[i];
  for (int i = tid; i < N_X; i += NTHREADS) { sbf3[i] = bf3[i]; sbg2[i] = bg2[i]; }

  // X0 = Inputs[:, N_S:, 0]
  for (int i = tid; i < MT * N_X; i += NTHREADS) {
    int r = i >> 6, j = i & 63;
    sX[i] = Inputs[(size_t)(b0 + r) * (D_IN * T_STEPS) + (size_t)(N_S + j) * T_STEPS];
  }
  __syncthreads();

  const int cM = (lane >> 4) << 3;   // C/D fragment row base for this lane
  const int cN = lane & 15;          // C/D fragment col within tile

  // ---------------- time loop ----------------------------------------------
  for (int t = 0; t < T_STEPS; ++t) {
    if (tid < MT * N_S) {
      int r = tid >> 2, c = tid & 3;
      sU[tid] = Inputs[(size_t)(b0 + r) * (D_IN * T_STEPS) + (size_t)c * T_STEPS + t];
      if (t + 1 < T_STEPS)
        __builtin_prefetch(&Inputs[(size_t)(b0 + r) * (D_IN * T_STEPS) +
                                   (size_t)c * T_STEPS + t + 1], 0, 1);
    }
    if (tid < MT * N_X / 16) {
      int r = tid >> 2, j = (tid & 3) * 16;
      if (t + 1 < T_STEPS)
        __builtin_prefetch(&noise[(size_t)(b0 + r) * (N_X * T_STEPS) +
                                  (size_t)j * T_STEPS + t + 1], 0, 1);
    }
#pragma unroll
    for (int i0 = 0; i0 < MT * N_X; i0 += NTHREADS) sAcc[i0 + tid] = 0.f;
    __syncthreads();

    // z = [X | u | 0-pad]  (shared by G and RK stage 0)
#pragma unroll
    for (int i0 = 0; i0 < E_Z; i0 += NTHREADS) {
      int i = i0 + tid;
      int r = i / KP, c = i - r * KP;
      float v = (c < N_X) ? sX[r * N_X + c]
                          : ((c < D_IN) ? sU[r * N_S + (c - N_X)] : 0.f);
      sZ[i] = (_Float16)v;
    }
    __syncthreads();

    // ========== G(X, u) ==========
    {   // G1 split-K: 8 tiles x 2 slices (K chunks {0,1} | {2}) over 16 waves
      const int nt = wave >> 1, ks = wave & 1;
      v8f c;
      if (ks == 0) c = gemm_tile<2>(sZ, KP, sWg1, KP / 32, nt, 0, lane);
      else         c = gemm_tile<1>(sZ, KP, sWg1, KP / 32, nt, 2, lane);
      *(v8f*)(sPart + wave * 256 + lane * 8) = c;
    }
    __syncthreads();
#pragma unroll
    for (int q0 = 0; q0 < 8 * 256; q0 += NTHREADS) {  // reduce -> tanh -> sHg
      int q = q0 + tid;
      int nt = q >> 8, p = q & 255;
      float v = sPart[(nt * 2 + 0) * 256 + p] + sPart[(nt * 2 + 1) * 256 + p];
      int l = p >> 3, r = p & 7;
      int M = r + ((l >> 4) << 3), N = (nt << 4) + (l & 15);
      sHg[M * H_G + N] = (_Float16)tanh_fast(v + sbg1[N]);
    }
    __syncthreads();
    {   // G2 split-K: 4 tiles x 4 slices (1 chunk each) over 16 waves
      const int nt = wave >> 2, ks = wave & 3;
      v8f c = gemm_tile<1>(sHg, H_G, sWg2, H_G / 32, nt, ks, lane);
      *(v8f*)(sPart + wave * 256 + lane * 8) = c;
    }
    __syncthreads();
#pragma unroll
    for (int q0 = 0; q0 < 4 * 256; q0 += NTHREADS) {  // reduce -> softplus -> sGv
      int q = q0 + tid;
      int nt = q >> 8, p = q & 255;
      float v = sPart[(nt * 4 + 0) * 256 + p] + sPart[(nt * 4 + 1) * 256 + p]
              + sPart[(nt * 4 + 2) * 256 + p] + sPart[(nt * 4 + 3) * 256 + p];
      int l = p >> 3, r = p & 7;
      int M = r + ((l >> 4) << 3), N = (nt << 4) + (l & 15);
      sGv[M * N_X + N] = SIGMA * softplus_fast(v + sbg2[N]);
    }
    __syncthreads();

    // ========== RK4 stages of F ==========
#pragma unroll 1
    for (int s = 0; s < 4; ++s) {
      const float aC = (s == 3) ? DT : 0.5f * DT;
      const float wC = (s == 1 || s == 2) ? 2.f : 1.f;

      if (s > 0) {   // stage 0 reuses the z built for G
#pragma unroll
        for (int i0 = 0; i0 < E_Z; i0 += NTHREADS) {
          int i = i0 + tid;
          int r = i / KP, c = i - r * KP;
          float v;
          if (c < N_X) v = sX[r * N_X + c] + aC * sK[r * N_X + c];
          else         v = (c < D_IN) ? sU[r * N_S + (c - N_X)] : 0.f;
          sZ[i] = (_Float16)v;
        }
      }
      __syncthreads();

      {   // L1: 16 tiles, one per wave (all waves busy)
        const int n0 = wave * 16;
        v8f c = gemm_tile<KP / 32>(sZ, KP, sWf1, KP / 32, wave, 0, lane);
#pragma unroll
        for (int r = 0; r < 8; ++r)
          sH1[(cM + r) * H_F + n0 + cN] = (_Float16)tanh_fast(c[r] + sbf1[n0 + cN]);
      }
      __syncthreads();
      {   // L2: 16 tiles, one per wave
        const int n0 = wave * 16;
        v8f c = gemm_tile<H_F / 32>(sH1, H_F, sWf2, H_F / 32, wave, 0, lane);
#pragma unroll
        for (int r = 0; r < 8; ++r)
          sH2[(cM + r) * H_F + n0 + cN] = (_Float16)tanh_fast(c[r] + sbf2[n0 + cN]);
      }
      __syncthreads();
      {   // L3 split-K: 4 tiles x 4 slices (2 chunks each) over 16 waves
        const int nt = wave >> 2, ks = wave & 3;
        v8f c = gemm_tile<2>(sH2, H_F, sWf3, H_F / 32, nt, ks * 2, lane);
        *(v8f*)(sPart + wave * 256 + lane * 8) = c;
      }
      __syncthreads();
#pragma unroll
      for (int q0 = 0; q0 < 4 * 256; q0 += NTHREADS) { // reduce -> K, Acc
        int q = q0 + tid;
        int nt = q >> 8, p = q & 255;
        float v = sPart[(nt * 4 + 0) * 256 + p] + sPart[(nt * 4 + 1) * 256 + p]
                + sPart[(nt * 4 + 2) * 256 + p] + sPart[(nt * 4 + 3) * 256 + p];
        int l = p >> 3, r = p & 7;
        int M = r + ((l >> 4) << 3), N = (nt << 4) + (l & 15);
        float kv = v + sbf3[N];
        int idx  = M * N_X + N;
        sK[idx] = kv;
        sAcc[idx] += wC * kv;
      }
      __syncthreads();
    }

    // ========== state update + output ==========
#pragma unroll
    for (int i0 = 0; i0 < MT * N_X; i0 += NTHREADS) {
      int i = i0 + tid;
      int r = i >> 6, j = i & 63;
      float dw = noise[(size_t)(b0 + r) * (N_X * T_STEPS) +
                       (size_t)j * T_STEPS + t] * SQRT_DT;
      float xn = sX[i] + (DT / 6.f) * sAcc[i] + sGv[i] * dw;
      sX[i] = xn;
      out[(size_t)(b0 + r) * (N_X * T_STEPS) + (size_t)j * T_STEPS + t] = xn;
    }
    __syncthreads();
  }
}

extern "C" void kernel_launch(void* const* d_in, const int* in_sizes, int n_in,
                              void* d_out, int out_size, void* d_ws, size_t ws_size,
                              hipStream_t stream) {
  const float* Inputs = (const float*)d_in[0];
  const float* noise  = (const float*)d_in[1];
  // d_in[2] = t0 (unused, zeros)
  const float* Wf1 = (const float*)d_in[3];
  const float* bf1 = (const float*)d_in[4];
  const float* Wf2 = (const float*)d_in[5];
  const float* bf2 = (const float*)d_in[6];
  const float* Wf3 = (const float*)d_in[7];
  const float* bf3 = (const float*)d_in[8];
  const float* Wg1 = (const float*)d_in[9];
  const float* bg1 = (const float*)d_in[10];
  const float* Wg2 = (const float*)d_in[11];
  const float* bg2 = (const float*)d_in[12];
  float* out = (float*)d_out;

  dim3 grid(B_TOT / MT);     // 16 persistent workgroups
  dim3 block(NTHREADS);      // 16 wave32 waves
  sde_rk4_kernel<<<grid, block, SMEM_BYTES, stream>>>(
      Inputs, noise, Wf1, bf1, Wf2, bf2, Wf3, bf3, Wg1, bg1, Wg2, bg2, out);
}
